// QuantumDeepField_29265907155795
// MI455X (gfx1250) — matrix operand order repaired and checked by reading, not compiled, not run on hardware
//
#include <hip/hip_runtime.h>
#include <hip/hip_bf16.h>

typedef __attribute__((ext_vector_type(16))) _Float16 v16h;
typedef __attribute__((ext_vector_type(8)))  _Float16 h8;
typedef __attribute__((ext_vector_type(8)))  float    v8f;

#define NB   8
#define NG   32768
#define NA   64
#define ND   128
#define EPSV 1e-12f

// pack two contiguous 8-half chunks into a v16h (A/B operand builder);
// works for both global (global_load_b128) and LDS (ds_load_b128) pointers.
static __device__ __forceinline__ v16h pack2(const _Float16* p0, const _Float16* p1) {
  v16h r;
  h8 a = *reinterpret_cast<const h8*>(p0);
  h8 b = *reinterpret_cast<const h8*>(p1);
#pragma unroll
  for (int j = 0; j < 8; ++j) { r[j] = a[j]; r[8 + j] = b[j]; }
  return r;
}

// ---------------- kernel 0: zero accumulators + convert MLP weights to f16 ----
__global__ void __launch_bounds__(256) prep_kernel(const float* __restrict__ Wfunc,
                                                   _Float16* __restrict__ Wf16,
                                                   float* gnorm2, float* monorm2,
                                                   float* final_acc) {
  int i = blockIdx.x * blockDim.x + threadIdx.x;
  if (i < 3 * ND * ND) Wf16[i] = (_Float16)Wfunc[i];
  if (i < NB * NA) gnorm2[i] = 0.f;
  if (i < NB * ND) { monorm2[i] = 0.f; final_acc[i] = 0.f; }
}

// ---------------- kernel 1: GTO basis + per-(b,a) sumsq over grid -------------
// block = 256 threads = 64 a-lanes x 4 g-lanes, 16 g iterations -> 64 g/block
__global__ void __launch_bounds__(256) gto_kernel(const float* __restrict__ dm,
                                                  const int* __restrict__ qn,
                                                  const int* __restrict__ ao,
                                                  const float* __restrict__ zeta,
                                                  _Float16* __restrict__ gto,
                                                  float* __restrict__ gnorm2) {
  __shared__ float red[256];
  int t = threadIdx.x;
  int a = t & 63, gl = t >> 6;
  int b = blockIdx.x >> 9;          // NG/64 = 512 blocks per molecule
  int gc = blockIdx.x & 511;
  int q = qn[b * NA + a];           // exponent d^(q) with q in {0,1,2}
  float z = zeta[ao[b * NA + a]];
  float local = 0.f;
  int gbase = gc * 64;
#pragma unroll 4
  for (int i = 0; i < 16; ++i) {
    int g = gbase + gl + i * 4;
    size_t idx = ((size_t)b * NG + g) * NA + a;
    __builtin_prefetch(&dm[idx + 4 * NA], 0, 1);   // gfx1250 global_prefetch_b8
    float d = dm[idx];
    float p = (q == 0) ? 1.0f : ((q == 1) ? d : d * d);
    float zd = z * d;
    float val = p * __expf(-zd * zd);
    gto[idx] = (_Float16)val;
    local += val * val;
  }
  red[t] = local;
  __syncthreads();
  if (t < 64) {
    float s = red[t] + red[t + 64] + red[t + 128] + red[t + 192];
    atomicAdd(&gnorm2[b * NA + t], s);
  }
}

// ---------------- kernel 2: fold all norms into transposed coeff matrix -------
// cT[b][d][a] = coef[ao][d] / (||c||_{a}(b,d) * ||gto||_{g}(b,a)), stored f16
__global__ void __launch_bounds__(128) cfold_kernel(const int* __restrict__ ao,
                                                    const float* __restrict__ coef,
                                                    const float* __restrict__ gnorm2,
                                                    _Float16* __restrict__ cT) {
  __shared__ int aoS[NA];
  __shared__ float gnS[NA];
  int b = blockIdx.x, t = threadIdx.x;    // t = d
  if (t < NA) {
    aoS[t] = ao[b * NA + t];
    gnS[t] = fmaxf(sqrtf(gnorm2[b * NA + t]), EPSV);
  }
  __syncthreads();
  float ss = 0.f;
  for (int a = 0; a < NA; ++a) { float c = coef[aoS[a] * ND + t]; ss += c * c; }
  float cn = fmaxf(sqrtf(ss), EPSV);
  for (int a = 0; a < NA; ++a) {
    float c = coef[aoS[a] * ND + t];
    cT[((size_t)(b * ND + t)) * NA + a] = (_Float16)(c / (cn * gnS[a]));
  }
}

// ---------------- kernel 3: WMMA GEMM  mo = gto @ cT^T, accumulate ||mo||² ----
// 8 waves/block, each wave: 16 grid rows x 128 dims (8 n-tiles, K=64 -> 2 wmma)
// B operand (cT, 16KB) staged once per block in LDS -> ds_load_b128 inner loop.
__global__ void __launch_bounds__(256) gemm1_kernel(const _Float16* __restrict__ gto,
                                                    const _Float16* __restrict__ cT,
                                                    _Float16* __restrict__ mo,
                                                    float* __restrict__ monorm2) {
  __shared__ __align__(16) _Float16 cs[ND * NA];      // 16 KB
  int t = threadIdx.x;
  int b = blockIdx.x >> 8;          // NG/128 = 256 tiles per molecule
  int gt = blockIdx.x & 255;

  // cooperative stage of this molecule's folded coefficient matrix
  {
    const h8* src = reinterpret_cast<const h8*>(cT + (size_t)b * ND * NA);
    h8* dst = reinterpret_cast<h8*>(cs);
#pragma unroll
    for (int i = 0; i < (ND * NA / 8) / 256; ++i) dst[t + i * 256] = src[t + i * 256];
  }
  __syncthreads();

  int wave = t >> 5, lane = t & 31;
  int ln = lane & 15, h = lane >> 4;
  int gbase = gt * 128 + wave * 16;

  const _Float16* arow = gto + ((size_t)(b * NG + gbase + ln)) * NA;
  v16h aLo = pack2(arow + h * 8,      arow + h * 8 + 16);       // K = 0..31
  v16h aHi = pack2(arow + 32 + h * 8, arow + 32 + h * 8 + 16);  // K = 32..63

#pragma unroll
  for (int nt = 0; nt < 8; ++nt) {
    const _Float16* brow = cs + (nt * 16 + ln) * NA;
    v16h bLo = pack2(brow + h * 8,      brow + h * 8 + 16);
    v16h bHi = pack2(brow + 32 + h * 8, brow + 32 + h * 8 + 16);
    v8f acc = {0.f, 0.f, 0.f, 0.f, 0.f, 0.f, 0.f, 0.f};
    acc = __builtin_amdgcn_wmma_f32_16x16x32_f16(false, aLo, false, bLo, (short)0, acc, false, false);
    acc = __builtin_amdgcn_wmma_f32_16x16x32_f16(false, aHi, false, bHi, (short)0, acc, false, false);
    float ls = 0.f;
#pragma unroll
    for (int r = 0; r < 8; ++r) {
      float v = acc[r];  // row = gbase + 8h + r, col = nt*16 + ln
      mo[((size_t)(b * NG + gbase + h * 8 + r)) * ND + nt * 16 + ln] = (_Float16)v;
      ls += v * v;
    }
    atomicAdd(&monorm2[b * ND + nt * 16 + ln], ls);
  }
}

// ---------------- kernel 4: density + 3-layer WMMA MLP + grid reduction -------
// All 3 layer weights (96 KB) + biases staged once per block into LDS; inner
// WMMA loop feeds A from registers and B from ds_load_b128. ~131 KB LDS of the
// WGP's 320 KB -> 2 concurrent workgroups per WGP.
__global__ void __launch_bounds__(256) mlp_kernel(const _Float16* __restrict__ mo,
                                                  const _Float16* __restrict__ Wf16,
                                                  const float* __restrict__ monorm2,
                                                  const float* __restrict__ Ne,
                                                  const float* __restrict__ Wpre_w,
                                                  const float* __restrict__ Wpre_b,
                                                  const float* __restrict__ Wfunc_b,
                                                  float* __restrict__ final_acc) {
  __shared__ float scl[ND], pw[ND], pb[ND];
  __shared__ float fb[3 * ND];
  __shared__ __align__(16) _Float16 wlds[3 * ND * ND];   // 96 KB: all layer weights
  __shared__ __align__(16) _Float16 vbuf[8][16 * ND];    // 32 KB: per-wave C->A tile
  int t = threadIdx.x;
  int b = blockIdx.x >> 8;
  int gt = blockIdx.x & 255;
  if (t < ND) {
    scl[t] = sqrtf(Ne[b] * (1.0f / ND)) / fmaxf(sqrtf(monorm2[b * ND + t]), EPSV);
    pw[t] = Wpre_w[t];
    pb[t] = Wpre_b[t];
  }
  if (t < 3 * ND - 256) fb[256 + t] = Wfunc_b[256 + t];
  fb[t] = Wfunc_b[t];
  // cooperative stage of all 3 layers' f16 weights: 6144 h8 chunks / 256 thr
  {
    const h8* src = reinterpret_cast<const h8*>(Wf16);
    h8* dst = reinterpret_cast<h8*>(wlds);
#pragma unroll
    for (int i = 0; i < (3 * ND * ND / 8) / 256; ++i) dst[t + i * 256] = src[t + i * 256];
  }
  __syncthreads();

  int wave = t >> 5, lane = t & 31;
  int ln = lane & 15, h = lane >> 4;
  int gbase = gt * 128 + wave * 16;

  // density for grid row ln: each half-wave handles 64 of the 128 dims
  const _Float16* mrow = mo + ((size_t)(b * NG + gbase + ln)) * ND + h * 64;
  float ds = 0.f;
#pragma unroll
  for (int j8 = 0; j8 < 8; ++j8) {
    h8 mv = *reinterpret_cast<const h8*>(mrow + j8 * 8);
#pragma unroll
    for (int j = 0; j < 8; ++j) {
      float x = (float)mv[j] * scl[h * 64 + j8 * 8 + j];
      ds += x * x;
    }
  }
  ds += __shfl_xor(ds, 16, 32);   // full 128-dim density

  // layer-1 input built directly in WMMA A layout: v0[k] = ds*W_pre[k]+b[k]
  v16h aF[4];
#pragma unroll
  for (int kk = 0; kk < 4; ++kk) {
#pragma unroll
    for (int j = 0; j < 8; ++j) {
      int k0 = kk * 32 + h * 8 + j;
      int k1 = k0 + 16;
      aF[kk][j]     = (_Float16)(ds * pw[k0] + pb[k0]);
      aF[kk][8 + j] = (_Float16)(ds * pw[k1] + pb[k1]);
    }
  }

  for (int l = 0; l < 3; ++l) {
    const _Float16* Wl = wlds + l * ND * ND;
    const float* bias = fb + l * ND;
#pragma unroll
    for (int nt = 0; nt < 8; ++nt) {
      v8f acc = {0.f, 0.f, 0.f, 0.f, 0.f, 0.f, 0.f, 0.f};
      const _Float16* wrow = Wl + (nt * 16 + ln) * ND;  // B col = out-neuron
#pragma unroll
      for (int kk = 0; kk < 4; ++kk) {
        v16h bF = pack2(wrow + kk * 32 + h * 8, wrow + kk * 32 + h * 8 + 16);
        acc = __builtin_amdgcn_wmma_f32_16x16x32_f16(false, aF[kk], false, bF, (short)0, acc, false, false);
      }
      float bb = bias[nt * 16 + ln];
      if (l < 2) {
#pragma unroll
        for (int r = 0; r < 8; ++r) {
          float x = fmaxf(acc[r] + bb, 0.f);
          vbuf[wave][(h * 8 + r) * ND + nt * 16 + ln] = (_Float16)x;
        }
      } else {  // last layer: sum over this wave's 16 grid rows, accumulate
        float ls = 0.f;
#pragma unroll
        for (int r = 0; r < 8; ++r) ls += fmaxf(acc[r] + bb, 0.f);
        atomicAdd(&final_acc[b * ND + nt * 16 + ln], ls);
      }
    }
    if (l < 2) {  // C-layout -> A-layout redistribution through LDS
      __syncthreads();
#pragma unroll
      for (int kk = 0; kk < 4; ++kk) {
        const _Float16* vp = &vbuf[wave][ln * ND + kk * 32];
        aF[kk] = pack2(vp + h * 8, vp + 16 + h * 8);
      }
      __syncthreads();
    }
  }
}

// ---------------- kernel 5: E[b] = final . W_prop_w + b ----------------------
__global__ void __launch_bounds__(128) out_kernel(const float* __restrict__ final_acc,
                                                  const float* __restrict__ Wp_w,
                                                  const float* __restrict__ Wp_b,
                                                  float* __restrict__ out) {
  __shared__ float red[128];
  int b = blockIdx.x, t = threadIdx.x;
  red[t] = final_acc[b * ND + t] * Wp_w[t];
  __syncthreads();
  for (int s = 64; s > 0; s >>= 1) {
    if (t < s) red[t] += red[t + s];
    __syncthreads();
  }
  if (t == 0) out[b] = red[0] + Wp_b[0];
}

extern "C" void kernel_launch(void* const* d_in, const int* in_sizes, int n_in,
                              void* d_out, int out_size, void* d_ws, size_t ws_size,
                              hipStream_t stream) {
  const float* dm      = (const float*)d_in[0];
  const int*   qn      = (const int*)d_in[1];
  const int*   ao      = (const int*)d_in[2];
  const float* Ne      = (const float*)d_in[3];
  const float* coef    = (const float*)d_in[4];
  const float* zeta    = (const float*)d_in[5];
  const float* Wpre_w  = (const float*)d_in[6];
  const float* Wpre_b  = (const float*)d_in[7];
  const float* Wfunc_w = (const float*)d_in[8];
  const float* Wfunc_b = (const float*)d_in[9];
  const float* Wprop_w = (const float*)d_in[10];
  const float* Wprop_b = (const float*)d_in[11];
  float* out = (float*)d_out;

  char* ws = (char*)d_ws;
  _Float16* gto      = (_Float16*)(ws);                 //  33,554,432 B
  _Float16* mo       = (_Float16*)(ws + 33554432);      //  67,108,864 B
  _Float16* cT       = (_Float16*)(ws + 100663296);     //     131,072 B
  _Float16* Wf16     = (_Float16*)(ws + 100794368);     //      98,304 B
  float*    gnorm2   = (float*)(ws + 100892672);        //       2,048 B
  float*    monorm2  = (float*)(ws + 100894720);        //       4,096 B
  float*    finalacc = (float*)(ws + 100898816);        //       4,096 B

  prep_kernel <<<192, 256, 0, stream>>>(Wfunc_w, Wf16, gnorm2, monorm2, finalacc);
  gto_kernel  <<<NB * (NG / 64), 256, 0, stream>>>(dm, qn, ao, zeta, gto, gnorm2);
  cfold_kernel<<<NB, 128, 0, stream>>>(ao, coef, gnorm2, cT);
  gemm1_kernel<<<NB * (NG / 128), 256, 0, stream>>>(gto, cT, mo, monorm2);
  mlp_kernel  <<<NB * (NG / 128), 256, 0, stream>>>(mo, Wf16, monorm2, Ne,
                                                    Wpre_w, Wpre_b, Wfunc_b, finalacc);
  out_kernel  <<<NB, 128, 0, stream>>>(finalacc, Wprop_w, Wprop_b, out);
}
